// CapsNet_54365696033057
// MI455X (gfx1250) — matrix operand order repaired and checked by reading, not compiled, not run on hardware
//
#include <hip/hip_runtime.h>
#include <hip/hip_bf16.h>

// ---------------------------------------------------------------------------
// CapsNet forward for MI455X (gfx1250, wave32, WMMA).
// Heavy op: PrimaryCaps conv as implicit GEMM 9216x256x(256*96pad) in f16 WMMA.
// v2: each block owns TWO 16-row M tiles; every B fragment is loaded once and
//     feeds two WMMAs (register-level B reuse -> 2x FLOP per L2 byte).
// ---------------------------------------------------------------------------

typedef _Float16 v16h __attribute__((ext_vector_type(16)));
typedef _Float16 v8h  __attribute__((ext_vector_type(8)));
typedef float    v8f  __attribute__((ext_vector_type(8)));

#define NB   256        // batch
#define NR   1152       // routes
#define NC   10         // capsules
#define NO   16         // out dim
#define NDI  8          // in dim
#define KPAD 96         // 81 -> 96 per input channel (multiple of 32)
#define NIC  256        // primary-caps input channels
#define EPS  1e-5f

// ---------------------------------------------------------------------------
// 1) Conv1: 1->256, k9 s1, +bias, ReLU; output stored f16 (A operand source)
//    h[((b*256+oc)*20+py)*20+px]
// ---------------------------------------------------------------------------
__global__ __launch_bounds__(256) void conv1_relu(
    const float* __restrict__ x, const float* __restrict__ cw,
    const float* __restrict__ cb, _Float16* __restrict__ h) {
  int idx = blockIdx.x * 256 + threadIdx.x;          // b*102400 + oc*400 + py*20 + px
  if (idx >= NB * 256 * 400) return;
  int px = idx % 20;
  int py = (idx / 20) % 20;
  int oc = (idx / 400) % 256;
  int b  = idx / (400 * 256);
  const float* xp = x + (size_t)b * 784 + py * 28 + px;
  const float* wp = cw + oc * 81;
  float acc = cb[oc];
#pragma unroll
  for (int ky = 0; ky < 9; ++ky)
#pragma unroll
    for (int kx = 0; kx < 9; ++kx)
      acc += xp[ky * 28 + kx] * wp[ky * 9 + kx];
  acc = fmaxf(acc, 0.f);
  h[idx] = (_Float16)acc;
}

// ---------------------------------------------------------------------------
// 2) Pack PrimaryCaps weights f32 (oc,ic,9,9) -> f16 wpad[oc][ic*96 + j],
//    j in [0,96), zero pad j>=81.
// ---------------------------------------------------------------------------
__global__ __launch_bounds__(256) void pack_w(
    const float* __restrict__ pw, _Float16* __restrict__ wpad) {
  int idx = blockIdx.x * 256 + threadIdx.x;          // oc*24576 + ic*96 + j
  if (idx >= 256 * NIC * KPAD) return;
  int j  = idx % KPAD;
  int ic = (idx / KPAD) % NIC;
  int oc = idx / (NIC * KPAD);
  _Float16 v = (_Float16)0.f;
  if (j < 81) v = (_Float16)pw[((size_t)oc * NIC + ic) * 81 + j];
  wpad[idx] = v;
}

// ---------------------------------------------------------------------------
// 3) PrimaryCaps implicit GEMM with WMMA.
//    M = b*36 + s (9216), N = oc (256), K = ic-blocks of 96 (pad of 81).
//    Block: 128 threads = 4 waves; each wave -> 4 oc tiles (64 oc) and
//    TWO M tiles (register B reuse). Grid = 9216/32 = 288 blocks.
//    A fragment layout (contiguous k order used here, same as B): lane L (g=L>>4, m=L&15):
//      halves[0..7]  = A[m][k0 + 16g .. +7]
//      halves[8..15] = A[m][k0 + 16g + 8 .. +7]
//    B fragment (32x16, lane n = col): halves[0..15] = B[k0 + 16g .. +15][n].
// ---------------------------------------------------------------------------
__global__ __launch_bounds__(128) void caps_gemm(
    const _Float16* __restrict__ h, const _Float16* __restrict__ wpad,
    const float* __restrict__ pc_b, float* __restrict__ u_pre) {
  __shared__ __align__(16) _Float16 As[32][KPAD];    // 6 KB (two M tiles)
  const int m0   = blockIdx.x * 32;                  // 288 blocks
  const int lane = threadIdx.x & 31;
  const int wave = threadIdx.x >> 5;                 // 0..3
  const int oc_base = wave * 64;
  const int mrow = lane & 15;
  const int g    = lane >> 4;

  v8f acc[2][4] = {};

  for (int ic = 0; ic < NIC; ++ic) {
    // stage A tiles: 32 rows x 96 halves (zero pad j>=81)
    for (int idx = threadIdx.x; idx < 32 * KPAD; idx += 128) {
      int row = idx / KPAD, j = idx % KPAD;
      _Float16 val = (_Float16)0.f;
      if (j < 81) {
        int m  = m0 + row;
        int b  = m / 36, s = m % 36;
        int oy = s / 6,  ox = s % 6;
        int ky = j / 9,  kx = j % 9;
        val = h[(((size_t)b * 256 + ic) * 20 + (2 * oy + ky)) * 20 + (2 * ox + kx)];
      }
      As[row][j] = val;
    }
    __syncthreads();

#pragma unroll
    for (int kk = 0; kk < KPAD; kk += 32) {
      union { v16h v; v8h h2[2]; } af0, af1;
      af0.h2[0] = *(const v8h*)&As[mrow][kk + g * 16];
      af0.h2[1] = *(const v8h*)&As[mrow][kk + g * 16 + 8];
      af1.h2[0] = *(const v8h*)&As[16 + mrow][kk + g * 16];
      af1.h2[1] = *(const v8h*)&As[16 + mrow][kk + g * 16 + 8];
#pragma unroll
      for (int tn = 0; tn < 4; ++tn) {
        int oc = oc_base + tn * 16 + (lane & 15);
        const v16h bfrag =
            *(const v16h*)(wpad + (size_t)oc * (NIC * KPAD) + ic * KPAD + kk + g * 16);
        acc[0][tn] = __builtin_amdgcn_wmma_f32_16x16x32_f16(
            false, af0.v, false, bfrag, (short)0, acc[0][tn], false, false);
        acc[1][tn] = __builtin_amdgcn_wmma_f32_16x16x32_f16(
            false, af1.v, false, bfrag, (short)0, acc[1][tn], false, false);
      }
    }
    __syncthreads();
  }

  // epilogue: C layout (ISA 7.12.2): VGPR j, lane L -> M = j + 8*(L>>4), N = L&15
#pragma unroll
  for (int mt = 0; mt < 2; ++mt) {
#pragma unroll
    for (int tn = 0; tn < 4; ++tn) {
      int oc = oc_base + tn * 16 + (lane & 15);
      float bias = pc_b[oc];
#pragma unroll
      for (int j = 0; j < 8; ++j) {
        int mi = m0 + mt * 16 + j + 8 * g;
        int b  = mi / 36, s = mi % 36;
        u_pre[(size_t)b * 9216 + oc * 36 + s] = acc[mt][tn][j] + bias;
      }
    }
  }
}

// ---------------------------------------------------------------------------
// 4) Squash over capsule dim 8: u[b,r,:] = squash(u_pre[b, 8r .. 8r+7])
// ---------------------------------------------------------------------------
__global__ __launch_bounds__(256) void squash_u(
    const float* __restrict__ upre, float* __restrict__ u) {
  int idx = blockIdx.x * 256 + threadIdx.x;          // b*1152 + r
  if (idx >= NB * NR) return;
  const float* p = upre + (size_t)idx * 8;
  float xv[8], sn = 0.f;
#pragma unroll
  for (int i = 0; i < 8; ++i) { xv[i] = p[i]; sn += xv[i] * xv[i]; }
  float sc = sn / (1.f + sn) / (sqrtf(sn) + EPS);
  float* q = u + (size_t)idx * 8;
#pragma unroll
  for (int i = 0; i < 8; ++i) q[i] = xv[i] * sc;
}

// ---------------------------------------------------------------------------
// 5) Wo[r,c,i] = sum_o W[r,c,o,i]   (for t = sum_o u_hat)
// ---------------------------------------------------------------------------
__global__ __launch_bounds__(256) void wo_kernel(
    const float* __restrict__ W, float* __restrict__ Wo) {
  int idx = blockIdx.x * 256 + threadIdx.x;          // rc*8 + i
  if (idx >= NR * NC * NDI) return;
  int i = idx % 8, rc = idx / 8;
  float s = 0.f;
  for (int o = 0; o < NO; ++o) s += W[((size_t)rc * NO + o) * NDI + i];
  Wo[idx] = s;
}

// ---------------------------------------------------------------------------
// 6) t[(r*10+c)*256 + b] = Wo[r,c,:] . u[b,r,:]
// ---------------------------------------------------------------------------
__global__ __launch_bounds__(256) void t_kernel(
    const float* __restrict__ u, const float* __restrict__ Wo,
    float* __restrict__ t) {
  int idx = blockIdx.x * 256 + threadIdx.x;          // (r*10+c)*256 + b
  if (idx >= NR * NC * NB) return;
  int b  = idx % NB;
  int rc = idx / NB;
  int r  = rc / NC;
  const float* wr = Wo + (size_t)rc * NDI;
  const float* ur = u + ((size_t)b * NR + r) * NDI;
  float s = 0.f;
#pragma unroll
  for (int i = 0; i < 8; ++i) s += wr[i] * ur[i];
  t[idx] = s;
}

// ---------------------------------------------------------------------------
// 7) zero-init b_ij
// ---------------------------------------------------------------------------
__global__ __launch_bounds__(256) void zero_f32(float* __restrict__ p, int n) {
  int idx = blockIdx.x * 256 + threadIdx.x;
  if (idx < n) p[idx] = 0.f;
}

// ---------------------------------------------------------------------------
// 8) softmax over routes r for each (c,o): cij[r,c,o] = softmax_r(bij[r,c,o])
//    one block per (c,o), layout [r*160 + c*16 + o]
// ---------------------------------------------------------------------------
__global__ __launch_bounds__(256) void softmax_r(
    const float* __restrict__ bij, float* __restrict__ cij) {
  __shared__ float red[256];
  int co  = blockIdx.x;                              // 0..159
  int tid = threadIdx.x;
  float m = -3.4e38f;
  for (int r = tid; r < NR; r += 256) m = fmaxf(m, bij[r * 160 + co]);
  red[tid] = m; __syncthreads();
  for (int s = 128; s > 0; s >>= 1) {
    if (tid < s) red[tid] = fmaxf(red[tid], red[tid + s]);
    __syncthreads();
  }
  m = red[0]; __syncthreads();
  float sum = 0.f;
  for (int r = tid; r < NR; r += 256) sum += expf(bij[r * 160 + co] - m);
  red[tid] = sum; __syncthreads();
  for (int s = 128; s > 0; s >>= 1) {
    if (tid < s) red[tid] += red[tid + s];
    __syncthreads();
  }
  float inv = 1.f / red[0];
  for (int r = tid; r < NR; r += 256) cij[r * 160 + co] = expf(bij[r * 160 + co] - m) * inv;
}

// ---------------------------------------------------------------------------
// 9) s/v step: s[b,c,o] = sum_r cij[r,c,o] * (W[r,c,o,:].u[b,r,:]),
//    v = elementwise squash(s). One block per batch b, 160 threads = (c,o).
//    Writes v to d_out (b,c,o) and transposed vT[(c*16+o)*256+b].
// ---------------------------------------------------------------------------
__global__ __launch_bounds__(160) void routing_s(
    const float* __restrict__ u, const float* __restrict__ W,
    const float* __restrict__ cij, float* __restrict__ v_out,
    float* __restrict__ vT) {
  __shared__ float us[NR * NDI];                     // 36 KB
  const int b = blockIdx.x;
  for (int i = threadIdx.x; i < NR * NDI; i += 160)
    us[i] = u[(size_t)b * NR * NDI + i];
  __syncthreads();
  const int c = threadIdx.x / NO;                    // 0..9
  const int o = threadIdx.x % NO;
  float s = 0.f;
  for (int r = 0; r < NR; ++r) {
    const float* wr = W + ((size_t)(r * NC + c) * NO + o) * NDI;
    const float* ur = us + r * NDI;
    float d = 0.f;
#pragma unroll
    for (int i = 0; i < 8; ++i) d += wr[i] * ur[i];
    s += cij[(r * NC + c) * NO + o] * d;
  }
  float sn = s * s;
  float vv = sn / (1.f + sn) * s / (sqrtf(sn) + EPS);
  v_out[(b * NC + c) * NO + o] = vv;
  vT[(c * NO + o) * NB + b]    = vv;
}

// ---------------------------------------------------------------------------
// 10) b_ij update: bij[r,c,o] += (1/B) * sum_b t[b,r,c] * v[b,c,o]
// ---------------------------------------------------------------------------
__global__ __launch_bounds__(256) void routing_b(
    const float* __restrict__ t, const float* __restrict__ vT,
    float* __restrict__ bij) {
  int idx = blockIdx.x * 256 + threadIdx.x;          // r*160 + c*16 + o
  if (idx >= NR * NC * NO) return;
  int o = idx % NO;
  int c = (idx / NO) % NC;
  int r = idx / 160;
  const float* tp = t + (size_t)(r * NC + c) * NB;
  const float* vp = vT + (size_t)(c * NO + o) * NB;
  float a = 0.f;
  for (int b = 0; b < NB; ++b) a += tp[b] * vp[b];
  bij[idx] += a * (1.f / (float)NB);
}

// ---------------------------------------------------------------------------
// launch
// ---------------------------------------------------------------------------
extern "C" void kernel_launch(void* const* d_in, const int* in_sizes, int n_in,
                              void* d_out, int out_size, void* d_ws, size_t ws_size,
                              hipStream_t stream) {
  const float* x      = (const float*)d_in[0];   // (256,1,28,28)
  const float* conv_w = (const float*)d_in[1];   // (256,1,9,9)
  const float* conv_b = (const float*)d_in[2];   // (256,)
  const float* pc_w   = (const float*)d_in[3];   // (256,256,9,9)
  const float* pc_b   = (const float*)d_in[4];   // (256,)
  const float* W      = (const float*)d_in[5];   // (1,1152,10,16,8)
  float* out = (float*)d_out;                    // (256,10,16)

  size_t off = 0;
  auto wsalloc = [&](size_t bytes) -> void* {
    void* p = (char*)d_ws + off;
    off += (bytes + 255) & ~(size_t)255;
    return p;
  };
  _Float16* h    = (_Float16*)wsalloc((size_t)NB * 256 * 400 * 2);        // 52.4 MB
  _Float16* wpad = (_Float16*)wsalloc((size_t)256 * NIC * KPAD * 2);      // 12.6 MB
  float* u_pre   = (float*)wsalloc((size_t)NB * NR * NDI * 4);            //  9.4 MB
  float* u       = (float*)wsalloc((size_t)NB * NR * NDI * 4);            //  9.4 MB
  float* Wo      = (float*)wsalloc((size_t)NR * NC * NDI * 4);
  float* t       = (float*)wsalloc((size_t)NR * NC * NB * 4);             // 11.8 MB
  float* bij     = (float*)wsalloc((size_t)NR * NC * NO * 4);
  float* cij     = (float*)wsalloc((size_t)NR * NC * NO * 4);
  float* vT      = (float*)wsalloc((size_t)NC * NO * NB * 4);

  // Conv1 + ReLU -> f16
  conv1_relu<<<(NB * 256 * 400 + 255) / 256, 256, 0, stream>>>(x, conv_w, conv_b, h);
  // Pack weights
  pack_w<<<(256 * NIC * KPAD + 255) / 256, 256, 0, stream>>>(pc_w, wpad);
  // PrimaryCaps implicit GEMM (WMMA), 2 M-tiles per block
  caps_gemm<<<(NB * 36) / 32, 128, 0, stream>>>(h, wpad, pc_b, u_pre);
  // Squash
  squash_u<<<(NB * NR + 255) / 256, 256, 0, stream>>>(u_pre, u);
  // Routing precompute
  wo_kernel<<<(NR * NC * NDI + 255) / 256, 256, 0, stream>>>(W, Wo);
  t_kernel<<<(NR * NC * NB + 255) / 256, 256, 0, stream>>>(u, Wo, t);
  zero_f32<<<(NR * NC * NO + 255) / 256, 256, 0, stream>>>(bij, NR * NC * NO);

  // 3 routing iterations
  for (int it = 0; it < 3; ++it) {
    softmax_r<<<NC * NO, 256, 0, stream>>>(bij, cij);
    routing_s<<<NB, 160, 0, stream>>>(u, W, cij, out, vT);
    if (it < 2)
      routing_b<<<(NR * NC * NO + 255) / 256, 256, 0, stream>>>(t, vT, bij);
  }
}
